// BahdanauAttention_60662118089278
// MI455X (gfx1250) — matrix operand order repaired and checked by reading, not compiled, not run on hardware
//
#include <hip/hip_runtime.h>

// ---------------- Problem constants (from reference setup) ----------------
#define Bb 8
#define Ss 256
#define Tt 128
#define Dd 512
#define Uu 512
#define NEGV (-1.0e9f)

typedef __attribute__((ext_vector_type(16))) _Float16 v16h;
typedef __attribute__((ext_vector_type(8)))  float    v8f;

union V16HU { uint4 u[2]; v16h v; };

__device__ __forceinline__ v16h load2x16B(const _Float16* lo, const _Float16* hi) {
  V16HU r;
  r.u[0] = *(const uint4*)lo;
  r.u[1] = *(const uint4*)hi;
  return r.v;
}

__device__ __forceinline__ float fast_tanhf(float x) {
  // tanh(x) = (e^{2x}-1)/(e^{2x}+1), e^{2x} = exp2(2*log2(e)*x)
  float xx = fminf(fmaxf(x, -9.0f), 9.0f);
  float e  = exp2f(xx * 2.8853900817779268f);
  return (e - 1.0f) * __builtin_amdgcn_rcpf(e + 1.0f);
}

// ---------------- Prep: f32 -> f16 copies / transposes ----------------
__global__ void prep_enc_k(const float* __restrict__ enc,
                           _Float16* __restrict__ encH,
                           _Float16* __restrict__ encT) {
  int idx = blockIdx.x * 256 + threadIdx.x;             // over B*S*D = 1,048,576
  int d = idx & (Dd - 1);
  int s = (idx >> 9) & (Ss - 1);
  int b = idx >> 17;
  float v = enc[idx];
  encH[idx] = (_Float16)v;
  encT[((size_t)b * Dd + d) * Ss + s] = (_Float16)v;    // [B,D,S] for context-GEMM B-matrix
}

__global__ void prep_dec_k(const float* __restrict__ dec, _Float16* __restrict__ decH) {
  int idx = blockIdx.x * 256 + threadIdx.x;             // over B*T*D = 524,288
  decH[idx] = (_Float16)dec[idx];
}

__global__ void prep_w_k(const float* __restrict__ W1, const float* __restrict__ W2,
                         _Float16* __restrict__ W1T, _Float16* __restrict__ W2T) {
  int idx = blockIdx.x * 256 + threadIdx.x;             // over U*D = 262,144
  int d = idx & (Dd - 1);
  int u = idx >> 9;
  W1T[idx] = (_Float16)W1[(size_t)d * Uu + u];          // [U,D]: K-contiguous columns
  W2T[idx] = (_Float16)W2[(size_t)d * Uu + u];
}

__global__ void reduce_wv_k(const float* __restrict__ Wv, float* __restrict__ sumOut) {
  __shared__ float sm[512];
  sm[threadIdx.x] = Wv[threadIdx.x];
  __syncthreads();
  for (int st = 256; st > 0; st >>= 1) {
    if ((int)threadIdx.x < st) sm[threadIdx.x] += sm[threadIdx.x + st];
    __syncthreads();
  }
  if (threadIdx.x == 0) sumOut[0] = sm[0];
}

// ---------------- WMMA GEMM, 2x2 register-blocked (32x32 out per wave) ----------------
//  C[b] = A[b](MxK, row-major) * BT[b](NxK, row-major)^T + bias
//  M, N must be multiples of 32; K multiple of 32.
__global__ void wmma_gemm_nt_k(const _Float16* __restrict__ A,
                               const _Float16* __restrict__ BT,
                               const float* __restrict__ bias,
                               float* __restrict__ C,
                               int K, int N,
                               unsigned strideA, unsigned strideB, unsigned strideC,
                               int tilesPerBatch, int ntiles32, int nwaves) {
  int gwave = (int)((blockIdx.x * blockDim.x + threadIdx.x) >> 5);
  if (gwave >= nwaves) return;                          // wave-uniform (EXEC stays all-ones)
  int lane = threadIdx.x & 31;
  int b    = gwave / tilesPerBatch;
  int t    = gwave % tilesPerBatch;
  int mt   = t / ntiles32;                              // 32-row block index
  int nt   = t % ntiles32;                              // 32-col block index
  int half = lane >> 4;
  int l16  = lane & 15;

  const _Float16* Arow0 = A  + (size_t)b * strideA + (size_t)(mt * 32 + l16) * K;
  const _Float16* Arow1 = Arow0 + (size_t)16 * K;
  const _Float16* Brow0 = BT + (size_t)b * strideB + (size_t)(nt * 32 + l16) * K;
  const _Float16* Brow1 = Brow0 + (size_t)16 * K;

  v8f c00 = {}, c01 = {}, c10 = {}, c11 = {};
  for (int kb = 0; kb < K; kb += 32) {
    int ka = kb + 8 * half;                             // A frag: {ka..ka+7, ka+16..ka+23}
    int kbb = kb + 16 * half;                           // B frag: same k order as A
    v16h a0 = load2x16B(Arow0 + ka, Arow0 + ka + 16);
    v16h a1 = load2x16B(Arow1 + ka, Arow1 + ka + 16);
    v16h b0 = load2x16B(Brow0 + ka, Brow0 + ka + 16);
    v16h b1 = load2x16B(Brow1 + ka, Brow1 + ka + 16);
    c00 = __builtin_amdgcn_wmma_f32_16x16x32_f16(false, a0, false, b0, (short)0, c00, false, false);
    c01 = __builtin_amdgcn_wmma_f32_16x16x32_f16(false, a0, false, b1, (short)0, c01, false, false);
    c10 = __builtin_amdgcn_wmma_f32_16x16x32_f16(false, a1, false, b0, (short)0, c10, false, false);
    c11 = __builtin_amdgcn_wmma_f32_16x16x32_f16(false, a1, false, b1, (short)0, c11, false, false);
  }

  int n0 = nt * 32 + l16;
  int n1 = n0 + 16;
  float bn0 = bias ? bias[n0] : 0.0f;
  float bn1 = bias ? bias[n1] : 0.0f;
  float* Cb = C + (size_t)b * strideC;
  size_t r0 = (size_t)(mt * 32 + 8 * half) * N;         // rows m..m+7 (vgpr r)
  size_t r1 = (size_t)(mt * 32 + 16 + 8 * half) * N;
#pragma unroll
  for (int r = 0; r < 8; ++r) {
    Cb[r0 + (size_t)r * N + n0] = c00[r] + bn0;
    Cb[r0 + (size_t)r * N + n1] = c01[r] + bn1;
    Cb[r1 + (size_t)r * N + n0] = c10[r] + bn0;
    Cb[r1 + (size_t)r * N + n1] = c11[r] + bn1;
  }
}

// ---------------- logits[b,t,s] = sum_u tanh(ep+dp)*Wv[u] + bv + (1-m)*NEG*sumWv ----------
// grid: (B, T/8, S/8); block 256 (8 waves).  8x8 logit tile per block; ep/dp tiles in LDS.
// Wave w owns t-row w (dp + Wv cached in registers); loops over the 8 s-rows.
__global__ void logits_k(const float* __restrict__ encProj,
                         const float* __restrict__ decProj,
                         const float* __restrict__ Wv,
                         const float* __restrict__ bvp,
                         const float* __restrict__ sumWvP,
                         const unsigned char* __restrict__ mask,
                         float* __restrict__ logits) {
  __shared__ float sEp[8 * Uu];       // 16 KB
  __shared__ float sDp[8 * Uu];       // 16 KB
  __shared__ float sWv[Uu];           //  2 KB
  int b  = blockIdx.x;
  int t0 = blockIdx.y * 8;
  int s0 = blockIdx.z * 8;

  const float* epg = encProj + ((size_t)b * Ss + s0) * Uu;
  const float* dpg = decProj + ((size_t)(b * Tt + t0)) * Uu;
  for (int i = threadIdx.x; i < 8 * Uu; i += 256) { sEp[i] = epg[i]; sDp[i] = dpg[i]; }
  for (int i = threadIdx.x; i < Uu; i += 256)     { sWv[i] = Wv[i]; }
  __syncthreads();

  int wave = threadIdx.x >> 5, lane = threadIdx.x & 31;
  float bv = bvp[0], sumWv = sumWvP[0];

  // cache this wave's dp row + Wv in registers (u = lane + 32*j)
  float dpv[16], wvv[16];
#pragma unroll
  for (int j = 0; j < 16; ++j) {
    int u = lane + 32 * j;
    dpv[j] = sDp[wave * Uu + u];
    wvv[j] = sWv[u];
  }

  for (int si = 0; si < 8; ++si) {
    const float* epr = &sEp[si * Uu];
    float acc = 0.0f;
#pragma unroll
    for (int j = 0; j < 16; ++j) {
      int u = lane + 32 * j;
      acc = fmaf(fast_tanhf(epr[u] + dpv[j]), wvv[j], acc);
    }
    for (int off = 16; off > 0; off >>= 1) acc += __shfl_xor(acc, off, 32);
    if (lane == 0) {
      int s = s0 + si;
      float m = mask[b * Ss + s] ? 1.0f : 0.0f;
      logits[((size_t)(b * Tt + t0 + wave)) * Ss + s] = acc + bv + (1.0f - m) * NEGV * sumWv;
    }
  }
}

// ---------------- softmax over S per (b,t); one wave per row ----------------
__global__ void softmax_k(const float* __restrict__ logits,
                          float* __restrict__ wout,          // f32 weights -> d_out
                          _Float16* __restrict__ probsH) {   // f16 copy for context GEMM
  int row  = blockIdx.x * 8 + (threadIdx.x >> 5);            // bt
  int lane = threadIdx.x & 31;
  const float* L = logits + (size_t)row * Ss;
  float v[8];
  float mx = -3.4e38f;
#pragma unroll
  for (int j = 0; j < 8; ++j) { v[j] = L[lane + 32 * j]; mx = fmaxf(mx, v[j]); }
  for (int off = 16; off > 0; off >>= 1) mx = fmaxf(mx, __shfl_xor(mx, off, 32));
  float sum = 0.0f;
#pragma unroll
  for (int j = 0; j < 8; ++j) { v[j] = exp2f((v[j] - mx) * 1.4426950408889634f); sum += v[j]; }
  for (int off = 16; off > 0; off >>= 1) sum += __shfl_xor(sum, off, 32);
  float inv = 1.0f / sum;
#pragma unroll
  for (int j = 0; j < 8; ++j) {
    float w = v[j] * inv;
    wout[(size_t)row * Ss + lane + 32 * j]   = w;
    probsH[(size_t)row * Ss + lane + 32 * j] = (_Float16)w;
  }
}

// ---------------- launch ----------------
extern "C" void kernel_launch(void* const* d_in, const int* in_sizes, int n_in,
                              void* d_out, int out_size, void* d_ws, size_t ws_size,
                              hipStream_t stream) {
  (void)in_sizes; (void)n_in; (void)out_size; (void)ws_size;
  const float* enc = (const float*)d_in[0];          // [B,S,D]
  const float* dec = (const float*)d_in[1];          // [B,T,D]
  const unsigned char* mask = (const unsigned char*)d_in[2]; // [B,S] bool
  const float* W1 = (const float*)d_in[3];
  const float* b1 = (const float*)d_in[4];
  const float* W2 = (const float*)d_in[5];
  const float* b2 = (const float*)d_in[6];
  const float* Wv = (const float*)d_in[7];           // [U,1]
  const float* bv = (const float*)d_in[8];           // [1]

  float* ctx_out = (float*)d_out;                            // [B,T,D]
  float* w_out   = (float*)d_out + (size_t)Bb * Tt * Dd;     // [B,T,S]

  // workspace carve-up (bytes)
  char* ws = (char*)d_ws;
  _Float16* encH    = (_Float16*)ws;                 ws += (size_t)Bb * Ss * Dd * 2; // 2 MB
  _Float16* encT    = (_Float16*)ws;                 ws += (size_t)Bb * Dd * Ss * 2; // 2 MB
  _Float16* decH    = (_Float16*)ws;                 ws += (size_t)Bb * Tt * Dd * 2; // 1 MB
  _Float16* W1T     = (_Float16*)ws;                 ws += (size_t)Uu * Dd * 2;      // 0.5 MB
  _Float16* W2T     = (_Float16*)ws;                 ws += (size_t)Uu * Dd * 2;      // 0.5 MB
  float*    encProj = (float*)ws;                    ws += (size_t)Bb * Ss * Uu * 4; // 4 MB
  float*    decProj = (float*)ws;                    ws += (size_t)Bb * Tt * Uu * 4; // 2 MB
  float*    logits  = (float*)ws;                    ws += (size_t)Bb * Tt * Ss * 4; // 1 MB
  _Float16* probsH  = (_Float16*)ws;                 ws += (size_t)Bb * Tt * Ss * 2; // 0.5 MB
  float*    sumWv   = (float*)ws;                    ws += 256;

  // prep
  prep_enc_k<<<(Bb * Ss * Dd) / 256, 256, 0, stream>>>(enc, encH, encT);
  prep_dec_k<<<(Bb * Tt * Dd) / 256, 256, 0, stream>>>(dec, decH);
  prep_w_k  <<<(Uu * Dd) / 256, 256, 0, stream>>>(W1, W2, W1T, W2T);
  reduce_wv_k<<<1, 512, 0, stream>>>(Wv, sumWv);

  // enc_proj = encH[2048x512] * W1T^T + b1  -> [2048x512] f32
  {
    int tiles = (Bb * Ss / 32) * (Uu / 32);          // 1024 waves (32x32 per wave)
    wmma_gemm_nt_k<<<tiles * 32 / 256, 256, 0, stream>>>(
        encH, W1T, b1, encProj, Dd, Uu, 0u, 0u, 0u, tiles, Uu / 32, tiles);
  }
  // dec_proj = decH[1024x512] * W2T^T + b2  -> [1024x512] f32
  {
    int tiles = (Bb * Tt / 32) * (Uu / 32);          // 512 waves
    wmma_gemm_nt_k<<<tiles * 32 / 256, 256, 0, stream>>>(
        decH, W2T, b2, decProj, Dd, Uu, 0u, 0u, 0u, tiles, Uu / 32, tiles);
  }

  // logits (tanh + Wv contraction), then softmax
  logits_k<<<dim3(Bb, Tt / 8, Ss / 8), 256, 0, stream>>>(
      encProj, decProj, Wv, bv, sumWv, mask, logits);
  softmax_k<<<(Bb * Tt) / 8, 256, 0, stream>>>(logits, w_out, probsH);

  // context[b] = probs[b][128x256] * encT[b]^T -> [128x512] f32 (batched)
  {
    int tilesPerBatch = (Tt / 32) * (Dd / 32);       // 64 waves per batch
    int nwaves = Bb * tilesPerBatch;                 // 512
    wmma_gemm_nt_k<<<nwaves * 32 / 256, 256, 0, stream>>>(
        probsH, encT, nullptr, ctx_out, Ss, Dd,
        (unsigned)(Tt * Ss), (unsigned)(Dd * Ss), (unsigned)(Tt * Dd),
        tilesPerBatch, Dd / 32, nwaves);
  }
}
